// AbLang2TransformerEncoder_80496277062507
// MI455X (gfx1250) — matrix operand order repaired and checked by reading, not compiled, not run on hardware
//
#include <hip/hip_runtime.h>
#include <hip/hip_bf16.h>
#include <math.h>

// ---------------------------------------------------------------------------
// AbLang2 transformer encoder layer for MI455X (gfx1250, wave32, WMMA).
// Heavy math via v_wmma_f32_16x16x32_bf16; GEMM tiles staged with CDNA5
// async global->LDS copies (ASYNCcnt) and double-buffered LDS.
// ---------------------------------------------------------------------------

#define D_   768
#define H_   12
#define HD_  64
#define FF_  3072
#define B_   8
#define N_   1024
#define M_   (B_ * N_)          // 8192 tokens
#define EPS_ 1e-5f

typedef __attribute__((ext_vector_type(8)))  __bf16 v8bf;
typedef __attribute__((ext_vector_type(16))) __bf16 v16bf;
typedef __attribute__((ext_vector_type(8)))  float  v8f;

static __device__ __forceinline__ v16bf cat16(v8bf lo, v8bf hi) {
  return __builtin_shufflevector(lo, hi, 0,1,2,3,4,5,6,7,8,9,10,11,12,13,14,15);
}
static __device__ __forceinline__ v8bf ld8(const __bf16* p) { return *(const v8bf*)p; }

static __device__ __forceinline__ v8f wmma_bf16(v16bf a, v16bf b, v8f c) {
  // (neg_a, A, neg_b, B, c_mod, C, reuse_a, reuse_b)
  return __builtin_amdgcn_wmma_f32_16x16x32_bf16(false, a, false, b, (short)0, c,
                                                 false, false);
}

// CDNA5 async global->LDS 16B copy (GLOBAL_LOAD_ASYNC_TO_LDS_B128, ASYNCcnt).
// LDS destination = low 32 bits of the generic shared-aperture pointer.
static __device__ __forceinline__ void async_ld16(const __bf16* g, __bf16* l) {
  unsigned lds = (unsigned)(unsigned long long)(uintptr_t)l;
  unsigned long long ga = (unsigned long long)(uintptr_t)g;
  asm volatile("global_load_async_to_lds_b128 %0, %1, off"
               :: "v"(lds), "v"(ga) : "memory");
}
static __device__ __forceinline__ void wait_async0() {
  asm volatile("s_wait_asynccnt 0" ::: "memory");
}

// ---------------------------------------------------------------------------
// elementwise f32 -> bf16 cast (weights)
// ---------------------------------------------------------------------------
__global__ void cast_f32_bf16_kernel(const float* __restrict__ src,
                                     __bf16* __restrict__ dst, int n) {
  int i = blockIdx.x * blockDim.x + threadIdx.x;
  if (i < n) dst[i] = (__bf16)src[i];
}

// ---------------------------------------------------------------------------
// LayerNorm over D=768 (one block per token), bf16 output for the GEMMs
// ---------------------------------------------------------------------------
__global__ __launch_bounds__(256) void ln_cast_kernel(
    const float* __restrict__ x, const float* __restrict__ w,
    const float* __restrict__ b, __bf16* __restrict__ out) {
  __shared__ float red[2][8];
  const int row = blockIdx.x;
  const float* xr = x + (size_t)row * D_;
  float v[3];
  float s = 0.f, s2 = 0.f;
#pragma unroll
  for (int i = 0; i < 3; ++i) {
    v[i] = xr[threadIdx.x + i * 256];
    s += v[i];
    s2 += v[i] * v[i];
  }
#pragma unroll
  for (int off = 16; off; off >>= 1) {
    s  += __shfl_xor(s, off);
    s2 += __shfl_xor(s2, off);
  }
  const int wid = threadIdx.x >> 5, lid = threadIdx.x & 31;
  if (lid == 0) { red[0][wid] = s; red[1][wid] = s2; }
  __syncthreads();
  if (threadIdx.x == 0) {
    float a = 0.f, a2 = 0.f;
#pragma unroll
    for (int i = 0; i < 8; ++i) { a += red[0][i]; a2 += red[1][i]; }
    red[0][0] = a; red[1][0] = a2;
  }
  __syncthreads();
  const float mean = red[0][0] * (1.f / D_);
  const float var  = red[1][0] * (1.f / D_) - mean * mean;
  const float inv  = rsqrtf(var + EPS_);
  __bf16* orow = out + (size_t)row * D_;
#pragma unroll
  for (int i = 0; i < 3; ++i) {
    const int c = threadIdx.x + i * 256;
    orow[c] = (__bf16)((v[i] - mean) * inv * w[c] + b[c]);
  }
}

// ---------------------------------------------------------------------------
// Generic tiled GEMM: out = epilogue((A[M,K] @ W[N,K]^T + bias[N]) * scale)
// 256 threads / 8 waves. 128x128 block, wave tile 32x64 (8 WMMAs/K-step),
// BK=32. Double-buffered LDS filled with async global->LDS B128 copies;
// one s_wait_asynccnt + one barrier per K-step instead of two round trips.
// LDS rows padded to 40 bf16 (80B, multiple of 16B) for B128 alignment.
// ---------------------------------------------------------------------------
#define GBM 128
#define GBN 128
#define LSTR 40

__global__ __launch_bounds__(256) void gemm_bf16_kernel(
    const __bf16* __restrict__ A, const __bf16* __restrict__ W,
    const float* __restrict__ bias, const float* __restrict__ res,
    float scale, int M, int N, int K, int do_gelu,
    float* __restrict__ outF, __bf16* __restrict__ outB) {
  __shared__ __bf16 As[2][GBM * LSTR];
  __shared__ __bf16 Bs[2][GBN * LSTR];

  const int tid  = threadIdx.x;
  const int lane = tid & 31;
  const int wave = tid >> 5;
  const int lrow = lane & 15;
  const int hi   = (lane >> 4) & 1;   // upper half of the wave
  const int wm   = wave >> 1;         // 4 wave-rows (32 rows each)
  const int wn   = wave & 1;          // 2 wave-cols (64 cols each)
  const int m0   = blockIdx.y * GBM;
  const int n0   = blockIdx.x * GBN;

  // staging: 128 rows x 32 bf16 each for A and B; thread t owns row t/2,
  // half-row (t&1)*16, i.e. 2x16B async copies per matrix.
  const int srow = tid >> 1, shalf = tid & 1;
  const __bf16* Ag = A + (size_t)(m0 + srow) * K + shalf * 16;
  const __bf16* Wg = W + (size_t)(n0 + srow) * K + shalf * 16;

  const v8f zero = {0.f, 0.f, 0.f, 0.f, 0.f, 0.f, 0.f, 0.f};
  v8f acc[2][4];
#pragma unroll
  for (int mt = 0; mt < 2; ++mt)
#pragma unroll
    for (int nt = 0; nt < 4; ++nt) acc[mt][nt] = zero;

  auto issue = [&](int buf, int k0) {
    __bf16* al = &As[buf][srow * LSTR + shalf * 16];
    __bf16* bl = &Bs[buf][srow * LSTR + shalf * 16];
    async_ld16(Ag + k0,     al);
    async_ld16(Ag + k0 + 8, al + 8);
    async_ld16(Wg + k0,     bl);
    async_ld16(Wg + k0 + 8, bl + 8);
  };

  const int ksteps = K >> 5;
  issue(0, 0);
  for (int ks = 0; ks < ksteps; ++ks) {
    wait_async0();       // this wave's async fills of buf are in LDS
    __syncthreads();     // every wave's fills are in LDS
    if (ks + 1 < ksteps) issue((ks + 1) & 1, (ks + 1) << 5);

    const int buf = ks & 1;
    // A fragment: lane<16 holds K {0..7,16..23}, lane>=16 holds {8..15,24..31}
    v16bf afr[2];
#pragma unroll
    for (int mt = 0; mt < 2; ++mt) {
      const __bf16* p = &As[buf][(wm * 32 + mt * 16 + lrow) * LSTR] + hi * 8;
      afr[mt] = cat16(ld8(p), ld8(p + 16));
    }
    // B fragment: lane<16 holds K {0..7,16..23}, lane>=16 holds {8..15,24..31} (W row-major)
    v16bf bfr[4];
#pragma unroll
    for (int nt = 0; nt < 4; ++nt) {
      const __bf16* p = &Bs[buf][(wn * 64 + nt * 16 + lrow) * LSTR] + hi * 8;
      bfr[nt] = cat16(ld8(p), ld8(p + 16));
    }
#pragma unroll
    for (int mt = 0; mt < 2; ++mt)
#pragma unroll
      for (int nt = 0; nt < 4; ++nt)
        acc[mt][nt] = wmma_bf16(afr[mt], bfr[nt], acc[mt][nt]);
  }

  // Epilogue. C layout: VGPR r -> row r (lanes 0-15) / row r+8 (lanes 16-31),
  // col = lane%16.
#pragma unroll
  for (int mt = 0; mt < 2; ++mt) {
#pragma unroll
    for (int nt = 0; nt < 4; ++nt) {
      const int gmb = m0 + wm * 32 + mt * 16 + hi * 8;
      const int gn  = n0 + wn * 64 + nt * 16 + lrow;
      const float bv = bias[gn];
#pragma unroll
      for (int r = 0; r < 8; ++r) {
        const int gm = gmb + r;
        float v = (acc[mt][nt][r] + bv) * scale;
        if (do_gelu) v = 0.5f * v * (1.f + erff(v * 0.70710678118654752f));
        const size_t o = (size_t)gm * N + gn;
        if (res)  v += res[o];
        if (outF) outF[o] = v;
        if (outB) outB[o] = (__bf16)v;
      }
    }
  }
}

// ---------------------------------------------------------------------------
// RoPE + cast: q/k f32 [B,N,D] (head-major cols) -> bf16 [B,H,N,HD]
// ---------------------------------------------------------------------------
__global__ void rope_cast_kernel(const float* __restrict__ q,
                                 const float* __restrict__ k,
                                 const float* __restrict__ freqs,
                                 __bf16* __restrict__ qo,
                                 __bf16* __restrict__ ko) {
  const int idx = blockIdx.x * blockDim.x + threadIdx.x;  // B*N*H*32 threads
  const int p = idx & 31;
  int t = idx >> 5;
  const int h = t % H_; t /= H_;
  const int n = t % N_;
  const int b = t / N_;
  const float ang = (float)n * freqs[p];
  const float c = cosf(ang), s = sinf(ang);
  const size_t ib = (size_t)(b * N_ + n) * D_ + h * HD_ + 2 * p;
  const size_t ob = ((size_t)(b * H_ + h) * N_ + n) * HD_ + 2 * p;
  {
    const float x1 = q[ib], x2 = q[ib + 1];
    qo[ob]     = (__bf16)(x1 * c - x2 * s);
    qo[ob + 1] = (__bf16)(x2 * c + x1 * s);
  }
  {
    const float x1 = k[ib], x2 = k[ib + 1];
    ko[ob]     = (__bf16)(x1 * c - x2 * s);
    ko[ob + 1] = (__bf16)(x2 * c + x1 * s);
  }
}

// ---------------------------------------------------------------------------
// V transpose + cast: f32 [B,N,D] -> bf16 [B,H,HD,N]
// ---------------------------------------------------------------------------
__global__ void vtrans_cast_kernel(const float* __restrict__ v,
                                   __bf16* __restrict__ vt) {
  const int idx = blockIdx.x * blockDim.x + threadIdx.x;  // B*H*HD*N
  const int n = idx % N_;
  int t = idx / N_;
  const int d = t % HD_; t /= HD_;
  const int h = t % H_;
  const int b = t / H_;
  vt[idx] = (__bf16)v[(size_t)(b * N_ + n) * D_ + h * HD_ + d];
}

// ---------------------------------------------------------------------------
// Fused flash attention. Grid (B*H, N/128). 8 waves, 16 query rows per wave.
// S = Q K^T via 2 WMMAs / 16-key subtile, online softmax in fp32, P tile
// round-trips through wave-private LDS to become an A-fragment, O += P V^T
// via 4 WMMAs per 32-key step. Q already carries the 1/HD scaling.
// ---------------------------------------------------------------------------
__global__ __launch_bounds__(256) void attn_kernel(
    const __bf16* __restrict__ Q, const __bf16* __restrict__ Kk,
    const __bf16* __restrict__ VT, const unsigned char* __restrict__ mask,
    __bf16* __restrict__ O) {
  __shared__ __bf16 Plds[8 * 16 * 40];  // per-wave 16x32 P tile (padded rows)

  const int b = blockIdx.x / H_;
  const int h = blockIdx.x % H_;
  const int wave = threadIdx.x >> 5;
  const int lane = threadIdx.x & 31;
  const int lrow = lane & 15;
  const int hi   = (lane >> 4) & 1;
  const int qrow = blockIdx.y * 128 + wave * 16;

  const size_t headQK = (size_t)(b * H_ + h) * N_;  // rows of Q/K for head
  const size_t headV  = (size_t)(b * H_ + h) * HD_; // rows of V^T for head

  v16bf aq[2];
#pragma unroll
  for (int kc = 0; kc < 2; ++kc) {
    const __bf16* p = Q + (headQK + qrow + lrow) * HD_ + kc * 32 + hi * 8;
    aq[kc] = cat16(ld8(p), ld8(p + 16));
  }

  const v8f zero = {0.f, 0.f, 0.f, 0.f, 0.f, 0.f, 0.f, 0.f};
  v8f oacc[4] = {zero, zero, zero, zero};
  float rmax[8], rsum[8];
#pragma unroll
  for (int r = 0; r < 8; ++r) { rmax[r] = -1e30f; rsum[r] = 0.f; }

  __bf16* pl = Plds + wave * 16 * 40;

  for (int m0 = 0; m0 < N_; m0 += 32) {
    // ---- scores: 16 rows x 32 keys (two 16-key C tiles) ----
    v8f s0 = zero, s1 = zero;
#pragma unroll
    for (int kc = 0; kc < 2; ++kc) {
      const __bf16* kp0 = Kk + (headQK + m0 + lrow) * HD_ + kc * 32 + hi * 8;
      const __bf16* kp1 = kp0 + 16 * HD_;
      s0 = wmma_bf16(aq[kc], cat16(ld8(kp0), ld8(kp0 + 16)), s0);
      s1 = wmma_bf16(aq[kc], cat16(ld8(kp1), ld8(kp1 + 16)), s1);
    }
    // padding mask (key-wise)
    const unsigned char mk0 = mask[b * N_ + m0 + lrow];
    const unsigned char mk1 = mask[b * N_ + m0 + 16 + lrow];
#pragma unroll
    for (int r = 0; r < 8; ++r) {
      if (mk0) s0[r] = -1e30f;
      if (mk1) s1[r] = -1e30f;
    }

    // ---- online softmax per row (row lives across 16 lanes of a half) ----
#pragma unroll
    for (int r = 0; r < 8; ++r) {
      float mx = fmaxf(s0[r], s1[r]);
#pragma unroll
      for (int off = 1; off < 16; off <<= 1) mx = fmaxf(mx, __shfl_xor(mx, off));
      const float mnew  = fmaxf(rmax[r], mx);
      const float alpha = __expf(rmax[r] - mnew);
      rmax[r] = mnew;
      const float e0 = __expf(s0[r] - mnew);
      const float e1 = __expf(s1[r] - mnew);
      float ps = e0 + e1;
#pragma unroll
      for (int off = 1; off < 16; off <<= 1) ps += __shfl_xor(ps, off);
      rsum[r] = rsum[r] * alpha + ps;
#pragma unroll
      for (int dt = 0; dt < 4; ++dt) oacc[dt][r] *= alpha;
      const int prow = r + hi * 8;
      pl[prow * 40 + lrow]      = (__bf16)e0;
      pl[prow * 40 + 16 + lrow] = (__bf16)e1;
    }

    // ---- O += P V^T ----
    {
      const __bf16* pa = pl + lrow * 40 + hi * 8;   // A-fragment of P (16x32)
      v16bf ap = cat16(ld8(pa), ld8(pa + 16));
#pragma unroll
      for (int dt = 0; dt < 4; ++dt) {
        const __bf16* vp = VT + (headV + dt * 16 + lrow) * N_ + m0 + hi * 8;
        oacc[dt] = wmma_bf16(ap, cat16(ld8(vp), ld8(vp + 16)), oacc[dt]);
      }
    }
  }

  // ---- normalize + write bf16 [B,N,D] (head-major cols) ----
#pragma unroll
  for (int dt = 0; dt < 4; ++dt) {
#pragma unroll
    for (int r = 0; r < 8; ++r) {
      const int row = qrow + r + hi * 8;
      const float v = oacc[dt][r] / rsum[r];
      O[(size_t)(b * N_ + row) * D_ + h * HD_ + dt * 16 + lrow] = (__bf16)v;
    }
  }
}

// ---------------------------------------------------------------------------
// launcher
// ---------------------------------------------------------------------------
extern "C" void kernel_launch(void* const* d_in, const int* in_sizes, int n_in,
                              void* d_out, int out_size, void* d_ws, size_t ws_size,
                              hipStream_t stream) {
  (void)in_sizes; (void)n_in; (void)out_size; (void)ws_size;
  const float* x     = (const float*)d_in[0];
  const unsigned char* pmask = (const unsigned char*)d_in[1];
  const float* wq = (const float*)d_in[2];  const float* bq = (const float*)d_in[3];
  const float* wk = (const float*)d_in[4];  const float* bk = (const float*)d_in[5];
  const float* wv = (const float*)d_in[6];  const float* bv = (const float*)d_in[7];
  const float* wo = (const float*)d_in[8];  const float* bo = (const float*)d_in[9];
  const float* w1 = (const float*)d_in[10]; const float* b1 = (const float*)d_in[11];
  const float* w2 = (const float*)d_in[12]; const float* b2 = (const float*)d_in[13];
  const float* ln1w = (const float*)d_in[14]; const float* ln1b = (const float*)d_in[15];
  const float* ln2w = (const float*)d_in[16]; const float* ln2b = (const float*)d_in[17];
  const float* freqs = (const float*)d_in[18];
  float* out = (float*)d_out;

  char* wsp = (char*)d_ws;
  auto alloc = [&](size_t bytes) -> char* {
    char* p = wsp;
    wsp += (bytes + 255) & ~(size_t)255;
    return p;
  };
  __bf16* wq_bf = (__bf16*)alloc((size_t)D_ * D_ * 2);
  __bf16* wk_bf = (__bf16*)alloc((size_t)D_ * D_ * 2);
  __bf16* wv_bf = (__bf16*)alloc((size_t)D_ * D_ * 2);
  __bf16* wo_bf = (__bf16*)alloc((size_t)D_ * D_ * 2);
  __bf16* w1_bf = (__bf16*)alloc((size_t)FF_ * D_ * 2);
  __bf16* w2_bf = (__bf16*)alloc((size_t)FF_ * D_ * 2);
  __bf16* h_bf  = (__bf16*)alloc((size_t)M_ * D_ * 2);   // LN1 out, reused for LN2
  float*  q_f   = (float*) alloc((size_t)M_ * D_ * 4);
  float*  k_f   = (float*) alloc((size_t)M_ * D_ * 4);
  float*  v_f   = (float*) alloc((size_t)M_ * D_ * 4);
  __bf16* q_bf  = (__bf16*)alloc((size_t)M_ * D_ * 2);   // [B,H,N,HD]
  __bf16* k_bf  = (__bf16*)alloc((size_t)M_ * D_ * 2);   // [B,H,N,HD]
  __bf16* vT_bf = (__bf16*)alloc((size_t)M_ * D_ * 2);   // [B,H,HD,N]
  __bf16* o_bf  = (__bf16*)alloc((size_t)M_ * D_ * 2);   // attention out
  float*  x1_f  = (float*) alloc((size_t)M_ * D_ * 4);   // post-attention residual
  __bf16* ff_bf = (__bf16*)alloc((size_t)M_ * FF_ * 2);  // GELU(FFN1)

  const dim3 blk(256);
  cast_f32_bf16_kernel<<<(D_*D_ + 255) / 256, blk, 0, stream>>>(wq, wq_bf, D_*D_);
  cast_f32_bf16_kernel<<<(D_*D_ + 255) / 256, blk, 0, stream>>>(wk, wk_bf, D_*D_);
  cast_f32_bf16_kernel<<<(D_*D_ + 255) / 256, blk, 0, stream>>>(wv, wv_bf, D_*D_);
  cast_f32_bf16_kernel<<<(D_*D_ + 255) / 256, blk, 0, stream>>>(wo, wo_bf, D_*D_);
  cast_f32_bf16_kernel<<<(FF_*D_ + 255) / 256, blk, 0, stream>>>(w1, w1_bf, FF_*D_);
  cast_f32_bf16_kernel<<<(FF_*D_ + 255) / 256, blk, 0, stream>>>(w2, w2_bf, FF_*D_);

  ln_cast_kernel<<<M_, blk, 0, stream>>>(x, ln1w, ln1b, h_bf);

  // QKV projections (Q folds SCALING * 1/sqrt(HD) = 1/64; commutes w/ RoPE)
  const dim3 gdD(D_ / GBN, M_ / GBM);
  gemm_bf16_kernel<<<gdD, blk, 0, stream>>>(h_bf, wq_bf, bq, nullptr,
      1.0f / 64.0f, M_, D_, D_, 0, q_f, nullptr);
  gemm_bf16_kernel<<<gdD, blk, 0, stream>>>(h_bf, wk_bf, bk, nullptr,
      1.0f, M_, D_, D_, 0, k_f, nullptr);
  gemm_bf16_kernel<<<gdD, blk, 0, stream>>>(h_bf, wv_bf, bv, nullptr,
      1.0f, M_, D_, D_, 0, v_f, nullptr);

  rope_cast_kernel<<<(B_*N_*H_*(HD_/2)) / 256, blk, 0, stream>>>(
      q_f, k_f, freqs, q_bf, k_bf);
  vtrans_cast_kernel<<<(M_*D_) / 256, blk, 0, stream>>>(v_f, vT_bf);

  attn_kernel<<<dim3(B_*H_, N_/128), blk, 0, stream>>>(q_bf, k_bf, vT_bf, pmask, o_bf);

  gemm_bf16_kernel<<<gdD, blk, 0, stream>>>(o_bf, wo_bf, bo, x,
      1.0f, M_, D_, D_, 0, x1_f, nullptr);

  ln_cast_kernel<<<M_, blk, 0, stream>>>(x1_f, ln2w, ln2b, h_bf);

  gemm_bf16_kernel<<<dim3(FF_/GBN, M_/GBM), blk, 0, stream>>>(h_bf, w1_bf, b1, nullptr,
      1.0f, M_, FF_, D_, 1 /*gelu*/, nullptr, ff_bf);
  gemm_bf16_kernel<<<gdD, blk, 0, stream>>>(ff_bf, w2_bf, b2, x1_f,
      1.0f, M_, D_, FF_, 0, out, nullptr);
}
